// MultiHeadSelfAttention_62405874811925
// MI455X (gfx1250) — matrix operand order repaired and checked by reading, not compiled, not run on hardware
//
#include <hip/hip_runtime.h>
#include <hip/hip_bf16.h>
#include <math.h>

#define NUM_HEADS 16

typedef __bf16 bf16;
typedef __attribute__((ext_vector_type(16))) __bf16 v16bf;
typedef __attribute__((ext_vector_type(8)))  float  v8f;

// GEMM tiling
#define BM 128
#define BN 128
#define BK 64
#define LDK 72   // BK+8: 144B row stride (16B aligned, odd*4 bank stride)

// Flash tiling
#define FK 32          // k-chunk
#define LDKS 136       // 128+8 (K tile row stride, elems)
#define LDVS 40        // 32+8  (V tile row stride, elems)

// ---------------------------------------------------------------------------
// CDNA5 async global->LDS copies (ASYNCcnt-tracked), with safe fallback.
// ---------------------------------------------------------------------------
#if defined(__has_builtin)
#  if __has_builtin(__builtin_amdgcn_global_load_async_to_lds_b128)
#    define HAVE_ASYNC_LDS 1
#  endif
#endif
#ifndef HAVE_ASYNC_LDS
#  define HAVE_ASYNC_LDS 0
#endif

#if HAVE_ASYNC_LDS
typedef int v4i_vs __attribute__((vector_size(16)));  // matches builtin's V4i
static __device__ __forceinline__ void cp16_async(const bf16* gp, bf16* lp) {
  __builtin_amdgcn_global_load_async_to_lds_b128(
      (__attribute__((address_space(1))) v4i_vs*)gp,
      (__attribute__((address_space(3))) v4i_vs*)lp, 0, 0);
}
#endif

static __device__ __forceinline__ void wait_async0() {
#if defined(__has_builtin) && __has_builtin(__builtin_amdgcn_s_wait_asynccnt)
  __builtin_amdgcn_s_wait_asynccnt(0);
#else
  asm volatile("s_wait_asynccnt 0" ::: "memory");
#endif
}

// ---------------------------------------------------------------------------
// DPP helpers (wave32; 16-lane reductions / neighbor exchange, no LDS)
// ---------------------------------------------------------------------------
template <int CTRL>
static __device__ __forceinline__ float dpp_mov_f32(float x) {
  return __int_as_float(
      __builtin_amdgcn_update_dpp(0, __float_as_int(x), CTRL, 0xF, 0xF, true));
}
// quad_perm(1,0,3,2)=0xB1  quad_perm(2,3,0,1)=0x4E  row_half_mirror=0x141  row_mirror=0x140
static __device__ __forceinline__ float row16_max(float x) {
  x = fmaxf(x, dpp_mov_f32<0xB1>(x));
  x = fmaxf(x, dpp_mov_f32<0x4E>(x));
  x = fmaxf(x, dpp_mov_f32<0x141>(x));
  x = fmaxf(x, dpp_mov_f32<0x140>(x));
  return x;
}
static __device__ __forceinline__ float row16_sum(float x) {
  x += dpp_mov_f32<0xB1>(x);
  x += dpp_mov_f32<0x4E>(x);
  x += dpp_mov_f32<0x141>(x);
  x += dpp_mov_f32<0x140>(x);
  return x;
}

// ---------------------------------------------------------------------------
// WMMA fragment helpers (contiguous k = 16*half + s in both A and B, wave32)
// ---------------------------------------------------------------------------
static __device__ __forceinline__ v16bf load_a_frag(const bf16* base, int lda) {
  int lane = threadIdx.x & 31;
  int half = lane >> 4;
  int row  = lane & 15;
  const bf16* rp = base + (size_t)row * lda;
  v16bf a;
#pragma unroll
  for (int i = 0; i < 8; ++i) {
    int kk = half * 16 + i * 2;
    a[2 * i]     = rp[kk];
    a[2 * i + 1] = rp[kk + 1];
  }
  return a;
}

// B fragment 32x16 where source is row-major [N, K] (B[k][n] = S[n][k]):
// per lane n = lane%16, k = half*16 .. +15 -> 16 contiguous bf16 (2 x b128).
static __device__ __forceinline__ v16bf load_bT_frag(const bf16* base, int ldw) {
  int lane = threadIdx.x & 31;
  int half = lane >> 4;
  int n    = lane & 15;
  const bf16* rp = base + (size_t)n * ldw + half * 16;
  v16bf b;
#pragma unroll
  for (int j = 0; j < 16; ++j) b[j] = rp[j];
  return b;
}

static __device__ __forceinline__ v8f wmma_bf16(v16bf a, v16bf b, v8f c) {
  return __builtin_amdgcn_wmma_f32_16x16x32_bf16(false, a, false, b, (short)0, c, false, false);
}

// ---------------------------------------------------------------------------
// fp32 -> bf16 conversion
// ---------------------------------------------------------------------------
__global__ void f32_to_bf16_kernel(const float* __restrict__ in, bf16* __restrict__ out, size_t n) {
  size_t stride = (size_t)gridDim.x * blockDim.x;
  for (size_t i = (size_t)blockIdx.x * blockDim.x + threadIdx.x; i < n; i += stride)
    out[i] = (bf16)in[i];
}

// ---------------------------------------------------------------------------
// LDS-tiled GEMM: C[M,N] = A[M,K] @ W[N,K]^T, bf16 in, fp32 accumulate.
// 256 thr = 8 waves (4x2), block tile 128x128, BK=64 double-buffered LDS.
// MODE 0: bf16; 1: bf16 + fused RoPE (head dim 128); 2: f32; 3: bf16
// transposed per batch -> Ct[b][N][s].
// ---------------------------------------------------------------------------
template <int MODE>
__global__ void __launch_bounds__(256) gemm_tiled(const bf16* __restrict__ A,
                                                  const bf16* __restrict__ W,
                                                  void* __restrict__ Cout,
                                                  const long long* __restrict__ pos,
                                                  int M, int N, int K, int s) {
  __shared__ bf16 As[2][BM][LDK];
  __shared__ bf16 Bs[2][BN][LDK];

  const int tid  = threadIdx.x;
  const int wid  = __builtin_amdgcn_readfirstlane(tid >> 5);
  const int lane = tid & 31;
  const int half = lane >> 4;
  const int ln   = lane & 15;
  const int wm   = wid >> 1;  // 0..3
  const int wn   = wid & 1;   // 0..1
  const int m0   = blockIdx.x * BM;
  const int n0   = blockIdx.y * BN;

  // Staging: each thread moves 32 bf16 (4 x b128) per matrix per stage.
  const int lrow = tid >> 1;
  const int lcol = (tid & 1) * 32;
  const bf16* Ag = A + (size_t)(m0 + lrow) * K + lcol;
  const bf16* Wg = W + (size_t)(n0 + lrow) * K + lcol;

#if HAVE_ASYNC_LDS
  auto stage_async = [&](int buf, int k0) {
    const bf16* pa = Ag + k0;
    const bf16* pw = Wg + k0;
#pragma unroll
    for (int j = 0; j < 4; ++j) {
      cp16_async(pa + 8 * j, &As[buf][lrow][lcol + 8 * j]);
      cp16_async(pw + 8 * j, &Bs[buf][lrow][lcol + 8 * j]);
    }
  };
#else
  uint4 ra[4], rb[4];
  auto gload = [&](int k0) {
    const bf16* pa = Ag + k0;
    const bf16* pw = Wg + k0;
#pragma unroll
    for (int j = 0; j < 4; ++j) {
      ra[j] = *(const uint4*)(pa + 8 * j);
      rb[j] = *(const uint4*)(pw + 8 * j);
    }
  };
  auto sstore = [&](int buf) {
#pragma unroll
    for (int j = 0; j < 4; ++j) {
      *(uint4*)&As[buf][lrow][lcol + 8 * j] = ra[j];
      *(uint4*)&Bs[buf][lrow][lcol + 8 * j] = rb[j];
    }
  };
#endif

  v8f acc[2][4] = {};
  const int nst = K / BK;

#if HAVE_ASYNC_LDS
  stage_async(0, 0);
  wait_async0();
  __syncthreads();
#else
  gload(0);
  sstore(0);
  __syncthreads();
#endif

  for (int ks = 0; ks < nst; ++ks) {
    int buf = ks & 1;
#if HAVE_ASYNC_LDS
    if (ks + 1 < nst) stage_async(buf ^ 1, (ks + 1) * BK);  // DMA overlaps compute
#else
    if (ks + 1 < nst) gload((ks + 1) * BK);
#endif
#pragma unroll
    for (int kk = 0; kk < BK; kk += 32) {
      v16bf af0 = load_a_frag(&As[buf][wm * 32][kk],      LDK);
      v16bf af1 = load_a_frag(&As[buf][wm * 32 + 16][kk], LDK);
#pragma unroll
      for (int t = 0; t < 4; ++t) {
        v16bf bfr = load_bT_frag(&Bs[buf][wn * 64 + t * 16][kk], LDK);
        acc[0][t] = wmma_bf16(af0, bfr, acc[0][t]);
        acc[1][t] = wmma_bf16(af1, bfr, acc[1][t]);
      }
    }
#if HAVE_ASYNC_LDS
    if (ks + 1 < nst) wait_async0();
#else
    if (ks + 1 < nst) sstore(buf ^ 1);
#endif
    __syncthreads();
  }

  // Epilogue
  const float lg = 0.14392246f;  // ln(10000)/64
#pragma unroll
  for (int f = 0; f < 2; ++f) {
#pragma unroll
    for (int t = 0; t < 4; ++t) {
#pragma unroll
      for (int i = 0; i < 8; ++i) {
        int   row = m0 + wm * 32 + f * 16 + i + 8 * half;
        int   col = n0 + wn * 64 + t * 16 + ln;
        float v   = acc[f][t][i];
        if (MODE == 2) {
          ((float*)Cout)[(size_t)row * N + col] = v;
        } else if (MODE == 1) {
          // Fused RoPE: pair partner (col^1) sits in lane ln^1 -> DPP xor1.
          float other = dpp_mov_f32<0xB1>(v);
          int   srow  = row % s;
          float posf  = (float)pos[srow];
          int   ip    = (col & 127) >> 1;
          float ang   = posf * __expf(-lg * (float)ip);
          float c, sn;
          __sincosf(ang, &sn, &c);
          float res = (col & 1) ? (v * c + other * sn) : (v * c - other * sn);
          ((bf16*)Cout)[(size_t)row * N + col] = (bf16)res;
        } else if (MODE == 3) {
          int bidx = row / s, srow = row % s;
          ((bf16*)Cout)[(size_t)bidx * N * s + (size_t)col * s + srow] = (bf16)v;
        } else {
          ((bf16*)Cout)[(size_t)row * N + col] = (bf16)v;
        }
      }
    }
  }
}

// ---------------------------------------------------------------------------
// Causal flash attention, block-cooperative K/V staging.
// Q,K: bf16 [b,s,dm]; Vt: bf16 [b][dm][s]; Y: bf16 [b,s,dm].
// Block = 256 thr = 8 waves; wave w owns q rows [q0blk + 16w, +16).
// K (32x128) and V (128x32) tiles double-buffered in LDS.
// grid.x = b*h, grid.y = s/128.
// ---------------------------------------------------------------------------
__global__ void __launch_bounds__(256) flash_attn_kernel(const bf16* __restrict__ Q,
                                                         const bf16* __restrict__ K,
                                                         const bf16* __restrict__ Vt,
                                                         bf16* __restrict__ Y,
                                                         int b, int s, int h, int d) {
  const int dm   = h * d;
  const int bh   = blockIdx.x;
  const int bidx = bh / h;
  const int hd   = bh % h;
  const int tid  = threadIdx.x;
  const int wid  = __builtin_amdgcn_readfirstlane(tid >> 5);
  const int lane = tid & 31;
  const int half = lane >> 4;
  const int ln   = lane & 15;
  const int q0blk = blockIdx.y * 128;
  const int q0    = q0blk + wid * 16;  // scalar (wid is readfirstlane'd)

  const bf16* Qb  = Q  + (size_t)bidx * s * dm + (size_t)hd * d;
  const bf16* Kb  = K  + (size_t)bidx * s * dm + (size_t)hd * d;
  const bf16* Vtb = Vt + (size_t)bidx * dm * s + (size_t)hd * d * s;

  __shared__ bf16 Ks[2][FK][LDKS];    // K chunk: [kpos][dcol]
  __shared__ bf16 Vs[2][128][LDVS];   // V chunk: [dcol][kpos]
  __shared__ bf16 plds[8][16][32];    // per-wave P relay (C -> A layout)

  // Staging maps (one chunk = 8KB per tile, 32B per thread per tile).
  const int krow = tid >> 3, kseg = (tid & 7) * 16;   // K: 32 rows x 256B
  const int vrow = tid >> 1, vseg = (tid & 1) * 16;   // V: 128 rows x 64B

#if HAVE_ASYNC_LDS
  auto stage_async = [&](int buf, int kc) {
    const bf16* kp = Kb + (size_t)(kc + krow) * dm + kseg;
    cp16_async(kp,     &Ks[buf][krow][kseg]);
    cp16_async(kp + 8, &Ks[buf][krow][kseg + 8]);
    const bf16* vp = Vtb + (size_t)vrow * s + kc + vseg;
    cp16_async(vp,     &Vs[buf][vrow][vseg]);
    cp16_async(vp + 8, &Vs[buf][vrow][vseg + 8]);
  };
#else
  uint4 rk0, rk1, rv0, rv1;
  auto gstage = [&](int kc) {
    const bf16* kp = Kb + (size_t)(kc + krow) * dm + kseg;
    rk0 = *(const uint4*)kp;
    rk1 = *(const uint4*)(kp + 8);
    const bf16* vp = Vtb + (size_t)vrow * s + kc + vseg;
    rv0 = *(const uint4*)vp;
    rv1 = *(const uint4*)(vp + 8);
  };
  auto sstage = [&](int buf) {
    *(uint4*)&Ks[buf][krow][kseg]     = rk0;
    *(uint4*)&Ks[buf][krow][kseg + 8] = rk1;
    *(uint4*)&Vs[buf][vrow][vseg]     = rv0;
    *(uint4*)&Vs[buf][vrow][vseg + 8] = rv1;
  };
#endif

  // Per-wave Q fragments (16 rows x d=128), loaded once from global.
  v16bf qf[4];
#pragma unroll
  for (int f = 0; f < 4; ++f)
    qf[f] = load_a_frag(Qb + (size_t)q0 * dm + f * 32, dm);

  v8f   out[8] = {};
  float mrow[8], lrow[8];
#pragma unroll
  for (int i = 0; i < 8; ++i) { mrow[i] = -1e30f; lrow[i] = 0.0f; }

  const float scale  = 1.0f / sqrtf((float)d);
  const int   kendW  = q0 + 16;      // scalar: this wave's causal bound
  const int   kendBk = q0blk + 128;  // block's bound

#if HAVE_ASYNC_LDS
  stage_async(0, 0);
  wait_async0();
  __syncthreads();
#else
  gstage(0);
  sstage(0);
  __syncthreads();
#endif

  for (int kc = 0; kc < kendBk; kc += FK) {
    int buf = (kc >> 5) & 1;
#if HAVE_ASYNC_LDS
    if (kc + FK < kendBk) stage_async(buf ^ 1, kc + FK);  // DMA overlaps compute
#else
    if (kc + FK < kendBk) gstage(kc + FK);
#endif

    if (kc < kendW) {  // scalar branch; EXEC stays all-1s for WMMA
      // S = Q(16x128) @ K^T -> two 16x16 f32 fragments (from LDS)
      v8f s0 = {}, s1 = {};
#pragma unroll
      for (int f = 0; f < 4; ++f) {
        v16bf kb0 = load_bT_frag(&Ks[buf][0][f * 32],  LDKS);
        s0 = wmma_bf16(qf[f], kb0, s0);
        v16bf kb1 = load_bT_frag(&Ks[buf][16][f * 32], LDKS);
        s1 = wmma_bf16(qf[f], kb1, s1);
      }

      // Causal mask: scalar-guarded, only the diagonal chunk pays for it.
      if (kc + FK - 1 > q0) {
#pragma unroll
        for (int i = 0; i < 8; ++i) {
          int row = q0 + i + 8 * half;
          if (kc + ln > row) s0[i] = -1e30f;
          if (kc + 16 + ln > row) s1[i] = -1e30f;
        }
      }

#pragma unroll
      for (int i = 0; i < 8; ++i) {
        float v0    = s0[i] * scale;
        float v1    = s1[i] * scale;
        float cm    = row16_max(fmaxf(v0, v1));
        float nm    = fmaxf(mrow[i], cm);
        float alpha = __expf(mrow[i] - nm);
        float p0    = __expf(v0 - nm);
        float p1    = __expf(v1 - nm);
        float ps    = row16_sum(p0 + p1);
        lrow[i] = lrow[i] * alpha + ps;
        mrow[i] = nm;
#pragma unroll
        for (int t = 0; t < 8; ++t) out[t][i] *= alpha;
        int mt = i + 8 * half;
        plds[wid][mt][ln]      = (bf16)p0;
        plds[wid][mt][16 + ln] = (bf16)p1;
      }

      // P back in A-layout; per-wave LDS ops are in-order, no barrier needed.
      v16bf pf = load_a_frag(&plds[wid][0][0], 32);

      // out(16x128) += P(16x32) @ V(32x128), V from LDS in bT layout.
#pragma unroll
      for (int t = 0; t < 8; ++t) {
        v16bf vb = load_bT_frag(&Vs[buf][t * 16][0], LDVS);
        out[t] = wmma_bf16(pf, vb, out[t]);
      }
    }

#if HAVE_ASYNC_LDS
    if (kc + FK < kendBk) wait_async0();
#else
    if (kc + FK < kendBk) sstage(buf ^ 1);
#endif
    __syncthreads();
  }

#pragma unroll
  for (int i = 0; i < 8; ++i) {
    float inv_l = 1.0f / lrow[i];
    int   row   = q0 + i + 8 * half;
    bf16* yr    = Y + (size_t)bidx * s * dm + (size_t)row * dm + (size_t)hd * d;
#pragma unroll
    for (int t = 0; t < 8; ++t) yr[t * 16 + ln] = (bf16)(out[t][i] * inv_l);
  }
}

// ---------------------------------------------------------------------------
// Host launcher
// ---------------------------------------------------------------------------
extern "C" void kernel_launch(void* const* d_in, const int* in_sizes, int n_in,
                              void* d_out, int out_size, void* d_ws, size_t ws_size,
                              hipStream_t stream) {
  const int b = 4, s = 2048, dm = 2048, h = NUM_HEADS, d = dm / h;
  const float*     x   = (const float*)d_in[0];
  const long long* pos = (const long long*)d_in[1];
  const float*     Wq  = (const float*)d_in[2];
  const float*     Wk  = (const float*)d_in[3];
  const float*     Wv  = (const float*)d_in[4];
  const float*     Wo  = (const float*)d_in[5];

  const size_t S = (size_t)b * s * dm;
  const size_t W = (size_t)dm * dm;

  bf16* Xb  = (bf16*)d_ws;
  bf16* Qw  = Xb + S;
  bf16* Kw  = Qw + S;
  bf16* Vt  = Kw + S;   // transposed V: [b][dm][s]
  bf16* Yw  = Vt + S;
  bf16* Wqb = Yw + S;
  bf16* Wkb = Wqb + W;
  bf16* Wvb = Wkb + W;
  bf16* Wob = Wvb + W;

  // 1) fp32 -> bf16
  f32_to_bf16_kernel<<<4096, 256, 0, stream>>>(x,  Xb,  S);
  f32_to_bf16_kernel<<<1024, 256, 0, stream>>>(Wq, Wqb, W);
  f32_to_bf16_kernel<<<1024, 256, 0, stream>>>(Wk, Wkb, W);
  f32_to_bf16_kernel<<<1024, 256, 0, stream>>>(Wv, Wvb, W);
  f32_to_bf16_kernel<<<1024, 256, 0, stream>>>(Wo, Wob, W);

  // 2) Projections (RoPE fused into Q/K; V stored transposed)
  const int M = b * s;  // 8192
  dim3 gg(M / BM, dm / BN);
  gemm_tiled<1><<<gg, 256, 0, stream>>>(Xb, Wqb, Qw, pos,     M, dm, dm, s);
  gemm_tiled<1><<<gg, 256, 0, stream>>>(Xb, Wkb, Kw, pos,     M, dm, dm, s);
  gemm_tiled<3><<<gg, 256, 0, stream>>>(Xb, Wvb, Vt, nullptr, M, dm, dm, s);

  // 3) Causal flash attention (block-cooperative LDS staging)
  flash_attn_kernel<<<dim3(b * h, s / 128), 256, 0, stream>>>(Qw, Kw, Vt, Yw, b, s, h, d);

  // 4) Output projection (fp32 out)
  gemm_tiled<2><<<gg, 256, 0, stream>>>(Yw, Wob, d_out, nullptr, M, dm, dm, s);
}
